// TransformerModel_6751688589574
// MI455X (gfx1250) — matrix operand order repaired and checked by reading, not compiled, not run on hardware
//
#include <hip/hip_runtime.h>

// ---------------------------------------------------------------------------
// Scaled Llama backbone (4 layers) + LoRA(q,v) + cls head for MI455X (gfx1250)
// Big matmuls -> V_WMMA_F32_16X16X32_BF16 (wave32), tiles staged into LDS by
// the Tensor Data Mover (tensor_load_to_lds + s_wait_tensorcnt), double
// buffered, global_prefetch_b8 lookahead, ds_add_f32 LDS atomics for LoRA.
// ---------------------------------------------------------------------------

typedef __attribute__((ext_vector_type(16))) __bf16 v16bf;
typedef __attribute__((ext_vector_type(8)))  __bf16 v8bf;
typedef __attribute__((ext_vector_type(8)))  float  v8f;
typedef __attribute__((ext_vector_type(4)))  unsigned int v4u;
typedef __attribute__((ext_vector_type(8)))  int    v8i;
typedef __attribute__((ext_vector_type(4)))  int    v4i;

constexpr int LYR  = 4;
constexpr int DIM  = 2048;
constexpr int NH   = 16;
constexpr int FF   = 8192;
constexpr int RNK  = 16;
constexpr int NC   = 3;
constexpr int HDIM = 128;   // DIM / NH
constexpr int SEQ  = 1024;
constexpr int BB   = 2;
constexpr int TT   = BB * SEQ;          // 2048 tokens
constexpr float ALPHA_OVER_R = 32.0f / RNK;   // 2.0
constexpr float EPSF = 1e-5f;
constexpr float NEGF = -1e9f;

// ---------------------------------------------------------------------------
// Generic batched GEMM: C[b] = alpha * A[b] (MxK, row-major, bf16)
//                               * B[b]^T (B stored NxK row-major, bf16)
//                               (+ C[b] if accumulate)
// M % 128 == 0, N % 128 == 0, K % 32 == 0 (true for all call sites here).
// ---------------------------------------------------------------------------
#define BM 128
#define BN 128
#define BK 32
#define LSTR 40   // LDS row stride in halves (32 data + 8 pad, 16B-aligned rows)

static __device__ inline v16bf make16(v8bf lo, v8bf hi) {
    v16bf r;
#pragma unroll
    for (int i = 0; i < 8; ++i) { r[i] = lo[i]; r[i + 8] = hi[i]; }
    return r;
}

// Issue one TDM 2D tile load: 128 rows x 32 halves from row-major [rows x K]
// global source into LDS with a 4-DWORD pad every 16 DWORDs (row stride 40
// halves == LSTR). Descriptor bitfields per CDNA5 ISA D# (CDNA5 ISA).
static __device__ inline void tdm_load_tile(const __bf16* gsrc,
                                            unsigned lds_byte_addr, int K)
{
    const unsigned long long ga = (unsigned long long)(uintptr_t)gsrc;

    v4u g0;
    g0[0] = 1u;                                   // count=1, user descriptor
    g0[1] = lds_byte_addr;                        // LDS destination (bytes)
    g0[2] = (unsigned)(ga & 0xffffffffu);         // global_addr[31:0]
    g0[3] = (unsigned)((ga >> 32) & 0x01ffffffu)  // global_addr[56:32]
          | (2u << 30);                           // type = 2 ("image")

    v8i g1;
    g1[0] = (1 << 16)      // data_size = 2 bytes
          | (1 << 20)      // pad_enable
          | (3 << 22)      // pad_interval = 16 DWORDs (one 32-half row)
          | (3 << 25);     // pad_amount   = 4 DWORDs  (8 halves -> LSTR=40)
    g1[1] = (K & 0xffff) << 16;                         // tensor_dim0[15:0]
    g1[2] = ((unsigned)K >> 16) | (BM << 16);           // dim0 hi | tensor_dim1=128
    g1[3] = (BK << 16);                                 // dim1 hi=0 | tile_dim0=32
    g1[4] = BM;                                         // tile_dim1=128, tile_dim2=0
    g1[5] = K;                                          // tensor_dim0_stride[31:0]
    g1[6] = 0;                                          // stride hi | dim1_stride lo
    g1[7] = 0;

    const v4i z4 = {0, 0, 0, 0};                        // groups 2/3 unused (2D)
    const v8i z8 = {0, 0, 0, 0, 0, 0, 0, 0};
    __builtin_amdgcn_tensor_load_to_lds(g0, g1, z4, z4, z8, 0);
}

__global__ __launch_bounds__(256)
void k_gemm_bf16(const __bf16* __restrict__ A, const __bf16* __restrict__ B,
                 float* __restrict__ C, int M, int N, int K,
                 long long sA, long long sB, long long sC,
                 float alpha, int accumulate)
{
    __shared__ __attribute__((aligned(16))) __bf16 As[2][BM * LSTR];
    __shared__ __attribute__((aligned(16))) __bf16 Bs[2][BN * LSTR];

    const int tid  = threadIdx.x;
    const int bz   = blockIdx.z;
    const __bf16* Ab = A + (long long)bz * sA;
    const __bf16* Bb = B + (long long)bz * sB;
    float*        Cb = C + (long long)bz * sC;

    const int m0 = blockIdx.y * BM;
    const int n0 = blockIdx.x * BN;

    // prefetch helpers (waves 1..7 prime GL2 two k-tiles ahead of the TDM)
    const int lrow  = tid >> 1;
    const int lpart = tid & 1;

    // wave geometry: 8 waves, 2 along M (64 rows) x 4 along N (32 cols)
    const int lane = tid & 31;
    const int wave = tid >> 5;
    const int wm   = wave & 1;
    const int wn   = wave >> 1;
    const int half = lane >> 4;      // lane group 0/1 (ISA K-split)
    const int l16  = lane & 15;

    v8f acc[4][2];
#pragma unroll
    for (int i = 0; i < 4; ++i)
#pragma unroll
        for (int j = 0; j < 2; ++j)
#pragma unroll
            for (int g = 0; g < 8; ++g) acc[i][j][g] = 0.0f;

    auto stage = [&](int buf, int kt) {
        if (wave == 0) {
            tdm_load_tile(Ab + (long long)m0 * K + kt * BK,
                          (unsigned)(uintptr_t)&As[buf][0], K);
            tdm_load_tile(Bb + (long long)n0 * K + kt * BK,
                          (unsigned)(uintptr_t)&Bs[buf][0], K);
        }
    };

    const int NT = K / BK;
    stage(0, 0);
    if (wave == 0) __builtin_amdgcn_s_wait_tensorcnt(0);
    __syncthreads();

    for (int kt = 0; kt < NT; ++kt) {
        const int cur = kt & 1;
        if (kt + 1 < NT) {
            stage(cur ^ 1, kt + 1);                       // TDM into next buffer
            if (wave != 0 && kt + 2 < NT) {               // global_prefetch_b8
                __builtin_prefetch(Ab + (long long)(m0 + lrow) * K + (kt + 2) * BK + lpart * 16, 0, 1);
                __builtin_prefetch(Bb + (long long)(n0 + lrow) * K + (kt + 2) * BK + lpart * 16, 0, 1);
            }
        }

        // A fragments: 16-bit 16x32 layout — lane group 0: K{0..7,16..23},
        // lane group 1: K{8..15,24..31}; chunks at half*8 and half*8+16.
        v16bf af[4];
#pragma unroll
        for (int i = 0; i < 4; ++i) {
            const int row = wm * 64 + i * 16 + l16;
            const __bf16* p = &As[cur][row * LSTR + half * 8];
            af[i] = make16(*(const v8bf*)p, *(const v8bf*)(p + 16));
        }
        // B fragments: 16-bit 32x16 layout — lane group 0: K{0..7,16..23},
        // lane group 1: K{8..15,24..31} (n = l16), matching A.
        v16bf bfr[2];
#pragma unroll
        for (int j = 0; j < 2; ++j) {
            const int col = wn * 32 + j * 16 + l16;
            const __bf16* p = &Bs[cur][col * LSTR + half * 8];
            bfr[j] = make16(*(const v8bf*)p, *(const v8bf*)(p + 16));
        }

#pragma unroll
        for (int i = 0; i < 4; ++i)
#pragma unroll
            for (int j = 0; j < 2; ++j)
                acc[i][j] = __builtin_amdgcn_wmma_f32_16x16x32_bf16(
                    false, af[i], false, bfr[j], (short)0, acc[i][j], false, false);

        if (wave == 0 && kt + 1 < NT) __builtin_amdgcn_s_wait_tensorcnt(0);
        __syncthreads();
    }

    // Epilogue: f32 C/D layout — VGPR g: M=g (lanes 0-15), M=g+8 (lanes 16-31)
#pragma unroll
    for (int i = 0; i < 4; ++i)
#pragma unroll
        for (int j = 0; j < 2; ++j)
#pragma unroll
            for (int g = 0; g < 8; ++g) {
                const int r = m0 + wm * 64 + i * 16 + half * 8 + g;
                const int c = n0 + wn * 32 + j * 16 + l16;
                const long long idx = (long long)r * N + c;
                float v = alpha * acc[i][j][g];
                if (accumulate) v += Cb[idx];
                Cb[idx] = v;
            }
}

// ---------------------------------------------------------------------------
// Elementwise / small kernels
// ---------------------------------------------------------------------------
__global__ __launch_bounds__(256)
void k_embed(const int* __restrict__ ids, const float* __restrict__ emb,
             float* __restrict__ x)
{
    const int t  = blockIdx.x;
    const long long id = ids[t];
    for (int e = threadIdx.x; e < DIM; e += 256)
        x[(long long)t * DIM + e] = emb[id * DIM + e];
}

__global__ __launch_bounds__(256)
void k_rmsnorm_bf16(const float* __restrict__ x, const float* __restrict__ w,
                    __bf16* __restrict__ out)
{
    const int t = blockIdx.x;
    const float* xr = x + (long long)t * DIM;
    __shared__ float red[256];
    float ss = 0.0f;
    for (int e = threadIdx.x; e < DIM; e += 256) { float v = xr[e]; ss += v * v; }
    red[threadIdx.x] = ss;
    __syncthreads();
    for (int s = 128; s > 0; s >>= 1) {
        if (threadIdx.x < s) red[threadIdx.x] += red[threadIdx.x + s];
        __syncthreads();
    }
    const float scale = rsqrtf(red[0] / (float)DIM + EPSF);
    for (int e = threadIdx.x; e < DIM; e += 256)
        out[(long long)t * DIM + e] = (__bf16)(xr[e] * scale * w[e]);
}

__global__ __launch_bounds__(256)
void k_cvt_bf16(const float* __restrict__ s, __bf16* __restrict__ d, long long n)
{
    const long long base = (long long)blockIdx.x * 1024 + threadIdx.x;
#pragma unroll
    for (int j = 0; j < 4; ++j) {
        const long long i = base + j * 256;
        if (i < n) d[i] = (__bf16)s[i];
    }
}

// LoRA: out[t,:] += (alpha/r) * (h[t,:] @ A^T) @ B^T   (A:[R,D] f32, B:[D,R] f32)
__global__ __launch_bounds__(256)
void k_lora(const __bf16* __restrict__ h, const float* __restrict__ Am,
            const float* __restrict__ Bm, float* __restrict__ out)
{
    const int t = blockIdx.x;
    __shared__ float sacc[RNK];
    if (threadIdx.x < RNK) sacc[threadIdx.x] = 0.0f;
    __syncthreads();

    float part[RNK];
#pragma unroll
    for (int r = 0; r < RNK; ++r) part[r] = 0.0f;
    for (int d = threadIdx.x; d < DIM; d += 256) {
        const float hv = (float)h[(long long)t * DIM + d];
#pragma unroll
        for (int r = 0; r < RNK; ++r) part[r] += hv * Am[(long long)r * DIM + d];
    }
#pragma unroll
    for (int r = 0; r < RNK; ++r) atomicAdd(&sacc[r], part[r]);  // ds_add_f32
    __syncthreads();

    for (int e = threadIdx.x; e < DIM; e += 256) {
        float acc = 0.0f;
#pragma unroll
        for (int r = 0; r < RNK; ++r) acc += sacc[r] * Bm[(long long)e * RNK + r];
        out[(long long)t * DIM + e] += ALPHA_OVER_R * acc;
    }
}

// RoPE + head split (mode=1): dst[b,h,s,hd] bf16.  mode=0: v transposed dst[b,h,hd,s].
__global__ __launch_bounds__(256)
void k_rope_split(const float* __restrict__ src, __bf16* __restrict__ dst, int mode)
{
    const int t = blockIdx.x;
    const int b = t / SEQ;
    const int s = t - b * SEQ;
    for (int e = threadIdx.x; e < DIM; e += 256) {
        const int h  = e >> 7;
        const int hd = e & 127;
        if (mode) {
            const int i = hd & 63;
            const float ang = (float)s * __powf(10000.0f, -2.0f * (float)i / 128.0f);
            const float c = __cosf(ang), sn = __sinf(ang);
            const float x0 = src[(long long)t * DIM + e];
            const float xr = (hd < 64)
                ? -src[(long long)t * DIM + (h * 128 + hd + 64)]
                :  src[(long long)t * DIM + (h * 128 + hd - 64)];
            dst[(((long long)b * NH + h) * SEQ + s) * HDIM + hd] = (__bf16)(x0 * c + xr * sn);
        } else {
            dst[(((long long)b * NH + h) * HDIM + hd) * SEQ + s] =
                (__bf16)src[(long long)t * DIM + e];
        }
    }
}

// masked softmax over one score row; writes bf16 probabilities
__global__ __launch_bounds__(256)
void k_softmax(const float* __restrict__ sc, __bf16* __restrict__ pr,
               const int* __restrict__ amask)
{
    const int row = blockIdx.x;          // query index 0..SEQ-1
    const int bh  = blockIdx.y;          // 0..B*NH-1
    const int b   = bh / NH;
    const float* srow = sc + ((long long)bh * SEQ + row) * SEQ;
    __bf16*      prow = pr + ((long long)bh * SEQ + row) * SEQ;
    __shared__ float red[256];

    float vals[4];
    float mx = -3.4e38f;
#pragma unroll
    for (int j = 0; j < 4; ++j) {
        const int col = threadIdx.x + j * 256;
        float v = srow[col];
        if (col > row) v += NEGF;                       // causal
        if (amask[b * SEQ + col] == 0) v += NEGF;       // padding
        vals[j] = v;
        mx = fmaxf(mx, v);
    }
    red[threadIdx.x] = mx;
    __syncthreads();
    for (int s = 128; s > 0; s >>= 1) {
        if (threadIdx.x < s) red[threadIdx.x] = fmaxf(red[threadIdx.x], red[threadIdx.x + s]);
        __syncthreads();
    }
    mx = red[0];
    __syncthreads();

    float sum = 0.0f;
#pragma unroll
    for (int j = 0; j < 4; ++j) { vals[j] = __expf(vals[j] - mx); sum += vals[j]; }
    red[threadIdx.x] = sum;
    __syncthreads();
    for (int s = 128; s > 0; s >>= 1) {
        if (threadIdx.x < s) red[threadIdx.x] += red[threadIdx.x + s];
        __syncthreads();
    }
    const float inv = 1.0f / red[0];
#pragma unroll
    for (int j = 0; j < 4; ++j) {
        const int col = threadIdx.x + j * 256;
        prow[col] = (__bf16)(vals[j] * inv);
    }
}

// merge heads: oh[b,h,s,hd] f32 -> ob[t, h*HD+hd] bf16
__global__ __launch_bounds__(256)
void k_merge(const float* __restrict__ oh, __bf16* __restrict__ ob)
{
    const int t = blockIdx.x;
    const int b = t / SEQ;
    const int s = t - b * SEQ;
    for (int e = threadIdx.x; e < DIM; e += 256) {
        const int h = e >> 7, hd = e & 127;
        ob[(long long)t * DIM + e] =
            (__bf16)oh[(((long long)b * NH + h) * SEQ + s) * HDIM + hd];
    }
}

__global__ __launch_bounds__(256)
void k_silu_mul(const float* __restrict__ g, const float* __restrict__ u,
                __bf16* __restrict__ a, long long n)
{
    const long long base = (long long)blockIdx.x * 1024 + threadIdx.x;
#pragma unroll
    for (int j = 0; j < 4; ++j) {
        const long long i = base + j * 256;
        if (i < n) {
            const float gv = g[i];
            const float sv = gv / (1.0f + __expf(-gv));
            a[i] = (__bf16)(sv * u[i]);
        }
    }
}

// final rms + pooling + tanh MLP head; one block per batch element
__global__ __launch_bounds__(256)
void k_head(const float* __restrict__ x, const int* __restrict__ amask,
            const float* __restrict__ fnorm,
            const float* __restrict__ W1, const float* __restrict__ b1,
            const float* __restrict__ W2, const float* __restrict__ b2,
            float* __restrict__ out)
{
    const int b = blockIdx.x;
    __shared__ float red[256];
    __shared__ float pooled[DIM];
    __shared__ float h1[DIM];

    int cnt = 0;
    for (int s = threadIdx.x; s < SEQ; s += 256) cnt += amask[b * SEQ + s];
    red[threadIdx.x] = (float)cnt;
    __syncthreads();
    for (int s = 128; s > 0; s >>= 1) {
        if (threadIdx.x < s) red[threadIdx.x] += red[threadIdx.x + s];
        __syncthreads();
    }
    const int idx = (int)red[0] - 1;
    __syncthreads();

    const float* xr = x + ((long long)b * SEQ + idx) * DIM;
    float ss = 0.0f;
    for (int e = threadIdx.x; e < DIM; e += 256) { float v = xr[e]; ss += v * v; }
    red[threadIdx.x] = ss;
    __syncthreads();
    for (int s = 128; s > 0; s >>= 1) {
        if (threadIdx.x < s) red[threadIdx.x] += red[threadIdx.x + s];
        __syncthreads();
    }
    const float scale = rsqrtf(red[0] / (float)DIM + EPSF);
    for (int e = threadIdx.x; e < DIM; e += 256)
        pooled[e] = xr[e] * scale * fnorm[e];
    __syncthreads();

    for (int e = threadIdx.x; e < DIM; e += 256) {
        float d = b1[e];
        for (int k = 0; k < DIM; ++k) d += pooled[k] * W1[(long long)e * DIM + k];
        h1[e] = tanhf(d);
    }
    __syncthreads();

    if (threadIdx.x < NC) {
        float d = b2[threadIdx.x];
        for (int k = 0; k < DIM; ++k) d += h1[k] * W2[(long long)threadIdx.x * DIM + k];
        out[b * NC + threadIdx.x] = d;
    }
}

// ---------------------------------------------------------------------------
// Host orchestration (graph-capture safe: only launches on `stream`)
// ---------------------------------------------------------------------------
extern "C" void kernel_launch(void* const* d_in, const int* in_sizes, int n_in,
                              void* d_out, int out_size, void* d_ws, size_t ws_size,
                              hipStream_t stream)
{
    (void)in_sizes; (void)n_in; (void)out_size; (void)ws_size;

    const int*   ids    = (const int*)  d_in[0];
    const int*   amask  = (const int*)  d_in[1];
    const float* embed  = (const float*)d_in[2];
    const float* Wq     = (const float*)d_in[3];
    const float* Wk     = (const float*)d_in[4];
    const float* Wv     = (const float*)d_in[5];
    const float* Wo     = (const float*)d_in[6];
    const float* Aq     = (const float*)d_in[7];
    const float* Bq     = (const float*)d_in[8];
    const float* Av     = (const float*)d_in[9];
    const float* Bv     = (const float*)d_in[10];
    const float* Wg     = (const float*)d_in[11];
    const float* Wu     = (const float*)d_in[12];
    const float* Wd     = (const float*)d_in[13];
    const float* norm1  = (const float*)d_in[14];
    const float* norm2  = (const float*)d_in[15];
    const float* fnorm  = (const float*)d_in[16];
    const float* hW1    = (const float*)d_in[17];
    const float* hb1    = (const float*)d_in[18];
    const float* hW2    = (const float*)d_in[19];
    const float* hb2    = (const float*)d_in[20];
    float* out = (float*)d_out;

    // workspace bump allocator
    char* ws = (char*)d_ws;
    size_t off = 0;
    auto alloc = [&](size_t bytes) -> void* {
        void* p = ws + off;
        off += (bytes + 255) & ~(size_t)255;
        return p;
    };

    float*  x   = (float*) alloc((size_t)TT * DIM * 4);
    __bf16* hb  = (__bf16*)alloc((size_t)TT * DIM * 2);
    __bf16* wb  = (__bf16*)alloc((size_t)FF * DIM * 2);   // weight conversion buffer
    float*  qf  = (float*) alloc((size_t)TT * DIM * 4);
    float*  kf  = (float*) alloc((size_t)TT * DIM * 4);
    float*  vf  = (float*) alloc((size_t)TT * DIM * 4);
    __bf16* qh  = (__bf16*)alloc((size_t)TT * DIM * 2);   // [B,H,S,HD]
    __bf16* kh  = (__bf16*)alloc((size_t)TT * DIM * 2);   // [B,H,S,HD]
    __bf16* vt  = (__bf16*)alloc((size_t)TT * DIM * 2);   // [B,H,HD,S]
    float*  oh  = (float*) alloc((size_t)TT * DIM * 4);   // [B,H,S,HD]
    __bf16* ob  = (__bf16*)alloc((size_t)TT * DIM * 2);
    __bf16* pr  = (__bf16*)alloc((size_t)BB * NH * SEQ * SEQ * 2);
    float*  gf  = (float*) alloc((size_t)TT * FF * 4);
    float*  uf  = (float*) alloc((size_t)TT * FF * 4);
    __bf16* ab  = (__bf16*)alloc((size_t)TT * FF * 2);
    float*  sc  = gf;   // scores [B,H,S,S] f32 aliases gf+uf (disjoint lifetimes)

    const dim3 blk(256);
    const long long nDD = (long long)DIM * DIM;
    const long long nFD = (long long)FF * DIM;
    const int gDD = (int)((nDD + 1023) / 1024);
    const int gFD = (int)((nFD + 1023) / 1024);
    const float inv_sqrt_hd = 0.08838834764831845f;   // 1/sqrt(128)

    // GEMM grids
    const dim3 gProj(DIM / BN, TT / BM, 1);           // N=2048, M=2048
    const dim3 gFfnUp(FF / BN, TT / BM, 1);           // N=8192, M=2048
    const dim3 gScore(SEQ / BN, SEQ / BM, BB * NH);   // N=1024, M=1024, batch 32
    const dim3 gAttnV(HDIM / BN, SEQ / BM, BB * NH);  // N=128,  M=1024, batch 32

    k_embed<<<TT, blk, 0, stream>>>(ids, embed, x);

    for (int l = 0; l < LYR; ++l) {
        const float* Wq_l = Wq + (long long)l * nDD;
        const float* Wk_l = Wk + (long long)l * nDD;
        const float* Wv_l = Wv + (long long)l * nDD;
        const float* Wo_l = Wo + (long long)l * nDD;
        const float* Aq_l = Aq + (long long)l * RNK * DIM;
        const float* Bq_l = Bq + (long long)l * DIM * RNK;
        const float* Av_l = Av + (long long)l * RNK * DIM;
        const float* Bv_l = Bv + (long long)l * DIM * RNK;
        const float* Wg_l = Wg + (long long)l * nFD;
        const float* Wu_l = Wu + (long long)l * nFD;
        const float* Wd_l = Wd + (long long)l * nFD;

        // --- attention block ---
        k_rmsnorm_bf16<<<TT, blk, 0, stream>>>(x, norm1 + (long long)l * DIM, hb);

        k_cvt_bf16<<<gDD, blk, 0, stream>>>(Wq_l, wb, nDD);
        k_gemm_bf16<<<gProj, blk, 0, stream>>>(hb, wb, qf, TT, DIM, DIM, 0, 0, 0, 1.0f, 0);
        k_lora<<<TT, blk, 0, stream>>>(hb, Aq_l, Bq_l, qf);

        k_cvt_bf16<<<gDD, blk, 0, stream>>>(Wk_l, wb, nDD);
        k_gemm_bf16<<<gProj, blk, 0, stream>>>(hb, wb, kf, TT, DIM, DIM, 0, 0, 0, 1.0f, 0);

        k_cvt_bf16<<<gDD, blk, 0, stream>>>(Wv_l, wb, nDD);
        k_gemm_bf16<<<gProj, blk, 0, stream>>>(hb, wb, vf, TT, DIM, DIM, 0, 0, 0, 1.0f, 0);
        k_lora<<<TT, blk, 0, stream>>>(hb, Av_l, Bv_l, vf);

        k_rope_split<<<TT, blk, 0, stream>>>(qf, qh, 1);
        k_rope_split<<<TT, blk, 0, stream>>>(kf, kh, 1);
        k_rope_split<<<TT, blk, 0, stream>>>(vf, vt, 0);

        // scores = (q @ k^T) / sqrt(HD), per (b,h)
        k_gemm_bf16<<<gScore, blk, 0, stream>>>(
            qh, kh, sc, SEQ, SEQ, HDIM,
            (long long)SEQ * HDIM, (long long)SEQ * HDIM, (long long)SEQ * SEQ,
            inv_sqrt_hd, 0);

        {
            dim3 gsm(SEQ, BB * NH);
            k_softmax<<<gsm, blk, 0, stream>>>(sc, pr, amask);
        }

        // o = probs @ v  (B-matrix = v^T stored [HD,S])
        k_gemm_bf16<<<gAttnV, blk, 0, stream>>>(
            pr, vt, oh, SEQ, HDIM, SEQ,
            (long long)SEQ * SEQ, (long long)HDIM * SEQ, (long long)SEQ * HDIM,
            1.0f, 0);

        k_merge<<<TT, blk, 0, stream>>>(oh, ob);

        // x += o @ Wo^T  (residual folded into epilogue)
        k_cvt_bf16<<<gDD, blk, 0, stream>>>(Wo_l, wb, nDD);
        k_gemm_bf16<<<gProj, blk, 0, stream>>>(ob, wb, x, TT, DIM, DIM, 0, 0, 0, 1.0f, 1);

        // --- FFN block ---
        k_rmsnorm_bf16<<<TT, blk, 0, stream>>>(x, norm2 + (long long)l * DIM, hb);

        k_cvt_bf16<<<gFD, blk, 0, stream>>>(Wg_l, wb, nFD);
        k_gemm_bf16<<<gFfnUp, blk, 0, stream>>>(hb, wb, gf, TT, FF, DIM, 0, 0, 0, 1.0f, 0);

        k_cvt_bf16<<<gFD, blk, 0, stream>>>(Wu_l, wb, nFD);
        k_gemm_bf16<<<gFfnUp, blk, 0, stream>>>(hb, wb, uf, TT, FF, DIM, 0, 0, 0, 1.0f, 0);

        {
            const long long nTF = (long long)TT * FF;
            const int gTF = (int)((nTF + 1023) / 1024);
            k_silu_mul<<<gTF, blk, 0, stream>>>(gf, uf, ab, nTF);
        }

        // x += act @ Wd^T  (Wd is [D,F] row-major == B[N=D,K=F])
        k_cvt_bf16<<<gFD, blk, 0, stream>>>(Wd_l, wb, nFD);
        k_gemm_bf16<<<gProj, blk, 0, stream>>>(ab, wb, x, TT, DIM, FF, 0, 0, 0, 1.0f, 1);
    }

    k_head<<<BB, blk, 0, stream>>>(x, amask, fnorm, hW1, hb1, hW2, hb2, out);
}
